// RNN_82016695485055
// MI455X (gfx1250) — matrix operand order repaired and checked by reading, not compiled, not run on hardware
//
#include <hip/hip_runtime.h>
#include <hip/hip_bf16.h>
#include <math.h>

typedef __attribute__((ext_vector_type(16))) _Float16 v16h;
typedef __attribute__((ext_vector_type(8)))  _Float16 v8h;
typedef __attribute__((ext_vector_type(8)))  float    v8f;
typedef __attribute__((ext_vector_type(4)))  unsigned int u32x4;
typedef __attribute__((ext_vector_type(8)))  int      i32x8;
typedef __attribute__((ext_vector_type(4)))  int      i32x4;

// Problem dims (fixed by reference)
#define BB 128
#define TT 1024
#define II 64
#define HH 512
#define OO 64

#define DT_C 0.5f
#define NS_C 0.03535533905932738f  // sqrt(0.5)*0.05

#if __has_builtin(__builtin_amdgcn_tensor_load_to_lds)
#define USE_TDM 1
#else
#define USE_TDM 0
#endif

// ---------------------------------------------------------------------------
// WMMA fragment loaders (CDNA5 16x16x32 f16, wave32; contiguous k = 16*(lane/16) + s in A and B)
// ---------------------------------------------------------------------------
__device__ inline v16h load_a_f16(const _Float16* __restrict__ src, int ld,
                                  int m0, int k0, int lane) {
    const int m  = m0 + (lane & 15);
    const int kh = (lane >> 4) * 16;
    const _Float16* p = src + (size_t)m * ld + k0;
    v16h a;
#pragma unroll
    for (int j = 0; j < 8; ++j) {
        a[j]     = p[kh + j];        // K = k0 + kh + j
        a[8 + j] = p[kh + 8 + j];    // K = k0 + kh + 8 + j
    }
    return a;
}

// B (32x16, KxN) from packed layout: pk[(kc*N + n)*32 + k_in_chunk]
__device__ inline v16h load_b_pk(const _Float16* __restrict__ pk, int N,
                                 int kc, int n0, int lane) {
    const int n  = n0 + (lane & 15);
    const int kh = (lane >> 4) * 16;
    const _Float16* p = pk + ((size_t)(kc * N + n) * 32) + kh;
    v16h b;
#pragma unroll
    for (int j = 0; j < 16; ++j) b[j] = p[j];
    return b;
}

// ---------------------------------------------------------------------------
// Grid-wide sense-reversing barrier (32 co-resident blocks)
// ---------------------------------------------------------------------------
__device__ inline void grid_barrier(unsigned* cnt, unsigned* gen, unsigned nblk) {
    __syncthreads();
    if (threadIdx.x == 0) {
        __threadfence();
        unsigned g = atomicAdd(gen, 0u);
        if (atomicAdd(cnt, 1u) == nblk - 1u) {
            atomicExch(cnt, 0u);
            atomicAdd(gen, 1u);
        } else {
            while (atomicAdd(gen, 0u) == g) { __builtin_amdgcn_s_sleep(2); }
        }
        __threadfence();
    }
    __syncthreads();
}

// ---------------------------------------------------------------------------
// TDM: DMA a contiguous 16KB panel (16 rows x 512 f16) Global -> LDS.
// Descriptor per CDNA5 ISA ch.8: 1-row tile of 8192 2-byte elements.
// ---------------------------------------------------------------------------
#if USE_TDM
__device__ inline void tdm_load_panel(const _Float16* gsrc, void* lds) {
    const unsigned long long ga = (unsigned long long)(uintptr_t)gsrc;
    u32x4 g0;
    g0[0] = 1u;                                    // count=1, user descriptor
    g0[1] = (unsigned)(uintptr_t)lds;              // lds_addr (low 32 bits of flat)
    g0[2] = (unsigned)ga;                          // global_addr[31:0]
    g0[3] = (unsigned)(ga >> 32) | (2u << 30);     // global_addr[56:32] | type=2

    i32x8 g1 = {};
    g1[0] = (1 << 16);                             // data_size = 1 (2-byte elems)
    g1[1] = (8192 << 16);                          // tensor_dim0 = 8192 (bits 63:48)
    g1[2] = (1 << 16);                             // tensor_dim1 = 1 (bits 95:80)
    g1[3] = (8192 << 16);                          // tile_dim0 = 8192 (bits 127:112)
    g1[4] = 1;                                     // tile_dim1 = 1
    g1[5] = 8192;                                  // tensor_dim0_stride = 8192
    i32x4 g2 = {};
    i32x4 g3 = {};
    i32x8 g4 = {};
    __builtin_amdgcn_tensor_load_to_lds(g0, g1, g2, g3, g4, 0);
}
#endif

// ---------------------------------------------------------------------------
// Prep kernels
// ---------------------------------------------------------------------------
__global__ void pack_wi_k(const float* __restrict__ wi, _Float16* __restrict__ pk) {
    int i = blockIdx.x * 256 + threadIdx.x;            // (kc*512 + n)*32 + j, kc<2
    if (i >= 2 * HH * 32) return;
    int j = i & 31, n = (i >> 5) & (HH - 1), kc = i >> 14;
    pk[i] = (_Float16)wi[(size_t)(kc * 32 + j) * HH + n];
}

__global__ void pack_wrec_k(const float* __restrict__ wrec, _Float16* __restrict__ pk) {
    int i = blockIdx.x * 256 + threadIdx.x;            // (kc*512 + n)*32 + j, kc<16
    if (i >= 16 * HH * 32) return;
    int j = i & 31, n = (i >> 5) & (HH - 1), kc = i >> 14;
    pk[i] = (_Float16)wrec[(size_t)n * HH + (kc * 32 + j)];   // B[k][n] = wrec[n][k]
}

__global__ void pack_wo_k(const float* __restrict__ wo, _Float16* __restrict__ pk) {
    int i = blockIdx.x * 256 + threadIdx.x;            // (kc*64 + n)*32 + j, kc<16
    if (i >= 16 * OO * 32) return;
    int j = i & 31, n = (i >> 5) & (OO - 1), kc = i >> 11;
    pk[i] = (_Float16)wo[(size_t)(kc * 32 + j) * OO + n];
}

__global__ void init_state_k(const float* __restrict__ h0, _Float16* __restrict__ th,
                             unsigned* __restrict__ bar) {
    int i = blockIdx.x * 256 + threadIdx.x;            // 2 * 128 * 512 elements
    if (i < 2 * BB * HH) {
        int n = i & (HH - 1);
        th[i] = (_Float16)tanhf(h0[n]);
    }
    if (blockIdx.x == 0 && threadIdx.x < 8) bar[threadIdx.x] = 0u;
}

// ---------------------------------------------------------------------------
// Kernel 1: xw = input @ wi + brec   (M=131072, K=64, N=512) -> f32 workspace
// ---------------------------------------------------------------------------
__global__ void __launch_bounds__(256)
xw_gemm_k(const float* __restrict__ inp, const _Float16* __restrict__ pkWi,
          const float* __restrict__ brec, float* __restrict__ xw) {
    __shared__ _Float16 sA[16 * II];
    const int lane = threadIdx.x & 31, wave = threadIdx.x >> 5;
    const int tile = blockIdx.x * 8 + wave;
    const int m0 = (tile >> 5) * 16;   // same for all waves in block (8 | 32)
    const int n0 = (tile & 31) * 16;

    for (int idx = threadIdx.x; idx < 16 * II; idx += 256) {
        int r = idx >> 6, c = idx & 63;
        sA[idx] = (_Float16)inp[(size_t)(m0 + r) * II + c];
    }
    __syncthreads();

    v8f acc = {};
#pragma unroll
    for (int kc = 0; kc < 2; ++kc) {
        v16h a = load_a_f16(sA, II, 0, kc * 32, lane);
        v16h b = load_b_pk(pkWi, HH, kc, n0, lane);
        acc = __builtin_amdgcn_wmma_f32_16x16x32_f16(false, a, false, b,
                                                     (short)0, acc, false, false);
    }
    const int n = n0 + (lane & 15);
    const float bias = brec[n];
#pragma unroll
    for (int j = 0; j < 8; ++j) {
        int m = m0 + j + (lane >> 4) * 8;
        xw[(size_t)m * HH + n] = acc[j] + bias;
    }
}

// ---------------------------------------------------------------------------
// Kernel 2: persistent recurrent scan. 32 blocks x 8 waves, one 16x16 h-tile
// per wave (h in VGPRs); A panel DMA'd into LDS via the Tensor Data Mover;
// one grid barrier per timestep; double-buffered tanh state.
// ---------------------------------------------------------------------------
__global__ void __launch_bounds__(256)
rnn_scan_k(const _Float16* __restrict__ pkWrec, const _Float16* __restrict__ pkWo,
           const float* __restrict__ xw, const float* __restrict__ noise,
           const float* __restrict__ h0, _Float16* __restrict__ th,
           float* __restrict__ out, unsigned* __restrict__ bar) {
    __shared__ _Float16 sA[16 * HH];   // 16 KB A panel (tanh(h) rows)
    const int lane = threadIdx.x & 31, wave = threadIdx.x >> 5;
    const int tile = blockIdx.x * 8 + wave;
    const int m0 = (tile >> 5) * 16;   // batch-row base (same across block)
    const int n0 = (tile & 31) * 16;   // hidden-col base
    const int n  = n0 + (lane & 15);

    // readout tile owned by this block (32 blocks <-> 8x4 tiles of [128x64])
    const int om0 = (blockIdx.x >> 2) * 16;
    const int on0 = (blockIdx.x & 3) * 16;

    _Float16* th0 = th;
    _Float16* th1 = th + (size_t)BB * HH;

    v8f h;
#pragma unroll
    for (int j = 0; j < 8; ++j) h[j] = h0[n];

    for (int t = 0; t < TT; ++t) {
        const _Float16* thp = (t & 1) ? th1 : th0;   // read tanh(h_{t-1})
        _Float16*       thn = (t & 1) ? th0 : th1;   // write tanh(h_t)

        // ---- Stage A panel (16 rows x 512 f16, contiguous 16 KB) into LDS ----
        __syncthreads();                       // sA no longer read by prev iter
#if USE_TDM
        if (wave == 0) {
            tdm_load_panel(thp + (size_t)m0 * HH, (void*)sA);
            __builtin_amdgcn_s_wait_tensorcnt(0);
        }
#else
        {
            const v8h* src = (const v8h*)(thp + (size_t)m0 * HH);
            v8h* dst = (v8h*)sA;
            for (int idx = threadIdx.x; idx < (16 * HH) / 8; idx += 256)
                dst[idx] = src[idx];
        }
#endif
        __syncthreads();

        // ---- rec tile = tanh(h) @ wrec^T : K = 512, 16 WMMAs ----
        v8f acc = {};
#pragma unroll
        for (int kc = 0; kc < 16; ++kc) {
            v16h a = load_a_f16(sA, HH, 0, kc * 32, lane);
            v16h b = load_b_pk(pkWrec, HH, kc, n0, lane);
            acc = __builtin_amdgcn_wmma_f32_16x16x32_f16(false, a, false, b,
                                                         (short)0, acc, false, false);
        }

        // ---- h update + write tanh(h_new); prefetch next step's streams ----
#pragma unroll
        for (int j = 0; j < 8; ++j) {
            int bb = m0 + j + (lane >> 4) * 8;
            size_t idx = ((size_t)bb * TT + t) * HH + n;
            float rec = acc[j] + xw[idx];
            float hn  = (1.0f - DT_C) * h[j] + DT_C * rec + NS_C * noise[idx];
            h[j] = hn;
            thn[(size_t)bb * HH + n] = (_Float16)tanhf(hn);
            if (t + 1 < TT) {                  // hide HBM latency of next step
                __builtin_prefetch(&xw[idx + HH], 0, 1);
                __builtin_prefetch(&noise[idx + HH], 0, 1);
            }
        }

        grid_barrier(bar, bar + 1, gridDim.x);

        // ---- Readout: out_t = tanh(h_t) @ wo ; one 16x16 tile per block ----
        if (wave == 0) {
            v8f oacc = {};
#pragma unroll
            for (int kc = 0; kc < 16; ++kc) {
                v16h a = load_a_f16(thn, HH, om0, kc * 32, lane);
                v16h b = load_b_pk(pkWo, OO, kc, on0, lane);
                oacc = __builtin_amdgcn_wmma_f32_16x16x32_f16(false, a, false, b,
                                                              (short)0, oacc, false, false);
            }
            const int on = on0 + (lane & 15);
#pragma unroll
            for (int j = 0; j < 8; ++j) {
                int ob = om0 + j + (lane >> 4) * 8;
                out[((size_t)ob * TT + t) * OO + on] = oacc[j];
            }
        }
    }
}

// ---------------------------------------------------------------------------
// Launcher
// ---------------------------------------------------------------------------
extern "C" void kernel_launch(void* const* d_in, const int* in_sizes, int n_in,
                              void* d_out, int out_size, void* d_ws, size_t ws_size,
                              hipStream_t stream) {
    const float* inp   = (const float*)d_in[0];
    const float* wi    = (const float*)d_in[1];
    const float* wrec  = (const float*)d_in[2];
    const float* wo    = (const float*)d_in[3];
    const float* brec  = (const float*)d_in[4];
    const float* h0    = (const float*)d_in[5];
    const float* noise = (const float*)d_in[6];
    float* out = (float*)d_out;

    // Workspace layout (256B aligned)
    char* ws = (char*)d_ws;
    _Float16* pkWi   = (_Float16*)(ws);                       //  64 KB
    _Float16* pkWrec = (_Float16*)(ws + 65536);               // 512 KB
    _Float16* pkWo   = (_Float16*)(ws + 65536 + 524288);      //  64 KB
    _Float16* th     = (_Float16*)(ws + 655360);              // 256 KB (x2 buffers)
    unsigned* bar    = (unsigned*)(ws + 917504);              // 256 B
    float*    xw     = (float*)   (ws + 917760);              // 256 MB

    pack_wi_k  <<<(2  * HH * 32 + 255) / 256, 256, 0, stream>>>(wi,   pkWi);
    pack_wrec_k<<<(16 * HH * 32 + 255) / 256, 256, 0, stream>>>(wrec, pkWrec);
    pack_wo_k  <<<(16 * OO * 32 + 255) / 256, 256, 0, stream>>>(wo,   pkWo);
    init_state_k<<<(2 * BB * HH + 255) / 256, 256, 0, stream>>>(h0, th, bar);

    const int xw_blocks = (BB * TT / 16) * (HH / 16) / 8;     // 32768
    xw_gemm_k<<<xw_blocks, 256, 0, stream>>>(inp, pkWi, brec, xw);

    rnn_scan_k<<<32, 256, 0, stream>>>(pkWrec, pkWo, xw, noise, h0, th, out, bar);
}
